// NonLocalMeans_59536836657751
// MI455X (gfx1250) — matrix operand order repaired and checked by reading, not compiled, not run on hardware
//
#include <hip/hip_runtime.h>
#include <hip/hip_bf16.h>

typedef __attribute__((ext_vector_type(16))) _Float16 v16h;
typedef __attribute__((ext_vector_type(8)))  float    v8f;

namespace {
constexpr int kH = 512, kW = 512, kC = 3, kN = 2;
constexpr int TILE = 16;
constexpr int RS   = 5;                 // search radius (SWS = 11)
constexpr int RT   = 2;                 // template radius (TWS = 5)
constexpr int HALO = RS + RT;           // 7
constexpr int YDIM = TILE + 2 * HALO;   // 30 (y / x halo region)
constexpr int YP   = YDIM + 1;          // 31 padded pitch
constexpr int ZR   = TILE + 2 * RT;     // 20 z rows
constexpr int ZC   = TILE + 2 * RT;     // 20 z cols
constexpr int ZP   = ZC + 1;            // 21 padded pitch
// exp(-d/3) == exp2(-d * log2(e)/3)
constexpr float NEG_LOG2E_OVER_H = -0.48089834696298783f;
}

__global__ __launch_bounds__(32)
void nlm_wmma_kernel(const float* __restrict__ xin, float* __restrict__ out) {
  // One wave32 per 16x16 output tile.
  __shared__ float s_x[kC][YDIM][YP];   // raw x with +-7 halo (only +-5 used for accum)
  __shared__ float s_y[YDIM][YP];       // luminance with +-7 halo
  __shared__ float s_z[ZR][ZP];         // per-shift squared diffs (20x20)

  const int lane = threadIdx.x;         // 0..31
  const int tx0  = blockIdx.x * TILE;
  const int ty0  = blockIdx.y * TILE;
  const float* img = xin + (size_t)blockIdx.z * kC * kH * kW;

  // ---- Stage 1: load halo tile (circular wrap == jnp.roll semantics) ----
  for (int idx = lane; idx < kC * YDIM * YDIM; idx += 32) {
    int c = idx / (YDIM * YDIM);
    int r = idx - c * (YDIM * YDIM);
    int j = r / YDIM;
    int i = r - j * YDIM;
    int gr = (ty0 - HALO + j) & (kH - 1);
    int gc = (tx0 - HALO + i) & (kW - 1);
    s_x[c][j][i] = img[((size_t)c * kH + gr) * kW + gc];
  }
  __syncthreads();

  // ---- Stage 2: luminance of clipped x ----
  for (int idx = lane; idx < YDIM * YDIM; idx += 32) {
    int j = idx / YDIM, i = idx - j * YDIM;
    float r = fminf(fmaxf(s_x[0][j][i], 0.0f), 1.0f);
    float g = fminf(fmaxf(s_x[1][j][i], 0.0f), 1.0f);
    float b = fminf(fmaxf(s_x[2][j][i], 0.0f), 1.0f);
    s_y[j][i] = 0.299f * r + 0.587f * g + 0.114f * b;
  }
  __syncthreads();

  // ---- Constant A fragment: banded vertical 5-tap box matrix (16x32, f16) ----
  // A[m][k] = 1 iff m <= k <= m+4   (k = z-row index, zero-padded K>=20)
  // A operand, contiguous k order (same as B): M = lane%16; element e:
  //   K = e + (lane>=16 ? 16 : 0)
  v16h afrag;
  {
    const int m = lane & 15;
    #pragma unroll
    for (int e = 0; e < 16; ++e) {
      int v = e >> 1, h = e & 1;
      int k = e + ((lane >= 16) ? 16 : 0); (void)v; (void)h;
      afrag[e] = (_Float16)((k >= m && k <= m + 4) ? 1.0f : 0.0f);
    }
  }

  // This lane's fixed fragment coordinates:
  const int bn    = lane & 15;                 // B/D column  (N = lane%16)
  const int kbase = (lane < 16) ? 0 : 16;      // B K-range base
  const int mrow0 = (lane < 16) ? 0 : 8;       // D row base (M = v + mrow0)

  float accR[8], accG[8], accB[8], accW[8];
  #pragma unroll
  for (int v = 0; v < 8; ++v) { accR[v] = 0.f; accG[v] = 0.f; accB[v] = 0.f; accW[v] = 0.f; }

  // ---- Stage 3: loop over 121 search shifts ----
  for (int sxi = -RS; sxi <= RS; ++sxi) {
    for (int syi = -RS; syi <= RS; ++syi) {
      // z[j][i] = (y[q] - y[q - s])^2 over the 20x20 template-extended tile.
      // z row j <-> image row (tile + j - RT); LDS row = j + (HALO-RT).
      for (int idx = lane; idx < ZR * ZC; idx += 32) {
        int j = idx / ZC, i = idx - j * ZC;
        float a = s_y[j + (HALO - RT)][i + (HALO - RT)];
        float b = s_y[j + (HALO - RT) - syi][i + (HALO - RT) - sxi];
        float d = a - b;
        s_z[j][i] = d * d;
      }
      __syncthreads();

      // B fragment = horizontal 5-tap sums H[k][n] (20x16, K zero-padded to 32).
      // B operand, contiguous k order (same as A): N = lane%16; element e: K = kbase + e.
      v16h bfrag;
      #pragma unroll
      for (int e = 0; e < 16; ++e) {
        int k = kbase + e;
        float hs = 0.0f;
        if (k < ZR) {
          hs = s_z[k][bn]     + s_z[k][bn + 1] + s_z[k][bn + 2]
             + s_z[k][bn + 3] + s_z[k][bn + 4];
        }
        bfrag[e] = (_Float16)hs;
      }

      // d2(16x16) = banded(16x32) x H(32x16): vertical 5-tap sum on the matrix pipe.
      v8f cz = {};
      v8f d2 = __builtin_amdgcn_wmma_f32_16x16x32_f16(
          /*neg_a=*/false, afrag, /*neg_b=*/false, bfrag,
          /*c_mod=*/(short)0, cz, /*reuse_a=*/false, /*reuse_b=*/false);

      // Weights + accumulation. D layout: lane holds (row = mrow0+v, col = bn).
      // Raw v_sqrt_f32 / v_exp_f32: inputs are non-negative, normal-range box
      // sums, so the hardware approximations are plenty ahead of the exp.
      #pragma unroll
      for (int v = 0; v < 8; ++v) {
        float dist = __builtin_amdgcn_sqrtf(d2[v]);
        float w    = __builtin_amdgcn_exp2f(dist * NEG_LOG2E_OVER_H);
        int   rj = mrow0 + v + HALO - syi;   // in [2,27]
        int   ri = bn        + HALO - sxi;   // in [2,27]
        accR[v] += w * s_x[0][rj][ri];
        accG[v] += w * s_x[1][rj][ri];
        accB[v] += w * s_x[2][rj][ri];
        accW[v] += w;
      }
      __syncthreads();   // s_z reused next shift
    }
  }

  // ---- Stage 4: normalize, clip, store ----
  float* o = out + (size_t)blockIdx.z * kC * kH * kW;
  #pragma unroll
  for (int v = 0; v < 8; ++v) {
    int gr = ty0 + mrow0 + v;
    int gc = tx0 + bn;
    float inv = 1.0f / accW[v];
    o[((size_t)0 * kH + gr) * kW + gc] = fminf(fmaxf(accR[v] * inv, 0.0f), 1.0f);
    o[((size_t)1 * kH + gr) * kW + gc] = fminf(fmaxf(accG[v] * inv, 0.0f), 1.0f);
    o[((size_t)2 * kH + gr) * kW + gc] = fminf(fmaxf(accB[v] * inv, 0.0f), 1.0f);
  }
}

extern "C" void kernel_launch(void* const* d_in, const int* in_sizes, int n_in,
                              void* d_out, int out_size, void* d_ws, size_t ws_size,
                              hipStream_t stream) {
  (void)in_sizes; (void)n_in; (void)out_size; (void)d_ws; (void)ws_size;
  const float* x = (const float*)d_in[0];
  float* out = (float*)d_out;
  dim3 grid(kW / TILE, kH / TILE, kN);   // 32 x 32 x 2 tiles
  dim3 block(32);                        // one wave32 per tile
  nlm_wmma_kernel<<<grid, block, 0, stream>>>(x, out);
}
